// Encoder_48644799595227
// MI455X (gfx1250) — matrix operand order repaired and checked by reading, not compiled, not run on hardware
//
#include <hip/hip_runtime.h>
#include <hip/hip_bf16.h>

typedef _Float16 half_t;
typedef __attribute__((ext_vector_type(16))) _Float16 v16h;
typedef __attribute__((ext_vector_type(4)))  _Float16 v4h;
typedef __attribute__((ext_vector_type(8)))  float    v8f;
typedef __attribute__((ext_vector_type(4)))  float    v4f;
typedef __attribute__((ext_vector_type(4)))  unsigned int v4u;

#define HS    136          // padded half-stride for LDS matrices (avoids 64-bank conflicts)
#define LN_EPS 1e-5f
#define RCLIP 0.015f
#define NW    8            // waves per block

// ---- LDS layouts (bytes), 8 waves => 128 rows ----
// w2t 128*136*2=34816 | w3t 34816 | hbuf 128*136*2=34816 | hbuf2 34816
#define OFF_W2T   0
#define OFF_W3T   34816
#define OFF_HB    69632
#define OFF_HB2   104448
#define OFF_FEAT  139264
// edge: feat 128*4*4=2048, w1 3*128*4=1536, b1 512
#define E_OFF_W1  141312
#define E_OFF_B1  142848
#define EDGE_LDS  143360
// node: feat 128*32*4=16384, w1 30*128*4=15360, b1 512
#define N_OFF_W1  155648
#define N_OFF_B1  171008
#define NODE_LDS  171520

__device__ __forceinline__ v16h load_frag(const half_t* p0, const half_t* p1) {
    union { v16h v; v4u q[2]; } u;
    u.q[0] = *(const v4u*)p0;
    u.q[1] = *(const v4u*)p1;
    return u.v;
}

// One 16x128 = A(16x128,f16) * W^T(128x128,f16) pass, f32 accumulate.
// Per K-chunk: batch all 16 B-fragment ds_load_b128 into independent registers,
// then fire 8 back-to-back v_wmma (one LDS wait per burst instead of per WMMA).
__device__ __forceinline__ void mma128(const half_t* abase, const half_t* wt,
                                       int lane, v8f acc[8]) {
    const int hi = (lane >> 4) & 1;
    const int c  = lane & 15;
    const half_t* wp = wt + c * HS + hi * 16;
    #pragma unroll
    for (int kc = 0; kc < 4; ++kc) {
        const int ka = kc * 32 + hi * 16;
        v16h a = load_frag(abase + ka, abase + ka + 8);
        v16h b[8];
        #pragma unroll
        for (int nt = 0; nt < 8; ++nt) {
            const half_t* bp = wp + nt * (16 * HS) + kc * 32;
            b[nt] = load_frag(bp, bp + 8);
        }
        #pragma unroll
        for (int nt = 0; nt < 8; ++nt)
            acc[nt] = __builtin_amdgcn_wmma_f32_16x16x32_f16(
                false, a, false, b[nt], (short)0, acc[nt], false, false);
    }
}

// -------------------- EDGE branch: 1e6 edges, feat(3)->128->128->128 + LN --------------------
__global__ void __launch_bounds__(256) edge_kernel(
    const float* __restrict__ pos, const int* __restrict__ nbr,
    const float* __restrict__ W1, const float* __restrict__ B1,
    const half_t* __restrict__ W2T, const float* __restrict__ B2,
    const half_t* __restrict__ W3T, const float* __restrict__ B3,
    const float* __restrict__ G, const float* __restrict__ BT,
    float* __restrict__ out, int numTiles)
{
    extern __shared__ char smem[];
    half_t* w2t   = (half_t*)(smem + OFF_W2T);
    half_t* w3t   = (half_t*)(smem + OFF_W3T);
    half_t* hbuf  = (half_t*)(smem + OFF_HB);
    half_t* hbuf2 = (half_t*)(smem + OFF_HB2);
    float*  feat  = (float*)(smem + OFF_FEAT);
    float*  w1l   = (float*)(smem + E_OFF_W1);
    float*  b1l   = (float*)(smem + E_OFF_B1);

    const int tid   = threadIdx.x;
    const int lane  = tid & 31;
    const int wid   = tid >> 5;
    const int wbase = wid * 16;
    const int c     = lane & 15;
    const int hi    = (lane >> 4) & 1;

    // stage transposed f16 weights into padded LDS
    for (int ci = tid; ci < 2048; ci += 256) {
        int n = ci >> 4, k8 = ci & 15;
        *(v4u*)(w2t + n * HS + k8 * 8) = *(const v4u*)(W2T + n * 128 + k8 * 8);
        *(v4u*)(w3t + n * HS + k8 * 8) = *(const v4u*)(W3T + n * 128 + k8 * 8);
    }
    for (int i = tid; i < 3 * 128; i += 256) w1l[i] = W1[i];
    if (tid < 128) b1l[tid] = B1[tid];
    __syncthreads();

    float b2v[8], b3v[8], gv[8], bv[8];
    #pragma unroll
    for (int nt = 0; nt < 8; ++nt) {
        int col = nt * 16 + c;
        b2v[nt] = B2[col]; b3v[nt] = B3[col]; gv[nt] = G[col]; bv[nt] = BT[col];
    }
    const int j0 = lane * 4;
    v4f w1x = *(const v4f*)(w1l + j0);
    v4f w1y = *(const v4f*)(w1l + 128 + j0);
    v4f w1d = *(const v4f*)(w1l + 256 + j0);
    v4f bb  = *(const v4f*)(b1l + j0);

    for (int T = blockIdx.x * NW + wid; T < numTiles; T += gridDim.x * NW) {
        // edge features (lanes 0..15, one edge each)
        if (lane < 16) {
            int e  = T * 16 + lane;
            int ri = nbr[e * 3 + 1], si = nbr[e * 3 + 2];
            float rx = pos[ri * 2]     - pos[si * 2];
            float ry = pos[ri * 2 + 1] - pos[si * 2 + 1];
            float dd = sqrtf(rx * rx + ry * ry);
            float* f = feat + (wbase + lane) * 4;
            f[0] = rx; f[1] = ry; f[2] = dd;
        }
        // layer1: 3 -> 128, relu, f16 to hbuf
        #pragma unroll
        for (int r = 0; r < 16; ++r) {
            const float* f = feat + (wbase + r) * 4;
            float rx = f[0], ry = f[1], dd = f[2];
            v4h pk;
            pk.x = (half_t)fmaxf(bb.x + rx * w1x.x + ry * w1y.x + dd * w1d.x, 0.f);
            pk.y = (half_t)fmaxf(bb.y + rx * w1x.y + ry * w1y.y + dd * w1d.y, 0.f);
            pk.z = (half_t)fmaxf(bb.z + rx * w1x.z + ry * w1y.z + dd * w1d.z, 0.f);
            pk.w = (half_t)fmaxf(bb.w + rx * w1x.w + ry * w1y.w + dd * w1d.w, 0.f);
            *(v4h*)(hbuf + (wbase + r) * HS + j0) = pk;
        }
        // layer2 (WMMA) + relu -> hbuf2
        v8f zero = {};
        v8f acc[8];
        #pragma unroll
        for (int nt = 0; nt < 8; ++nt) acc[nt] = zero;
        mma128(hbuf + (wbase + c) * HS, w2t, lane, acc);
        #pragma unroll
        for (int nt = 0; nt < 8; ++nt) {
            #pragma unroll
            for (int r = 0; r < 8; ++r) {
                float h = fmaxf(acc[nt][r] + b2v[nt], 0.f);
                hbuf2[(wbase + r + hi * 8) * HS + nt * 16 + c] = (half_t)h;
            }
        }
        // layer3 (WMMA) + bias + layernorm -> global
        #pragma unroll
        for (int nt = 0; nt < 8; ++nt) acc[nt] = zero;
        mma128(hbuf2 + (wbase + c) * HS, w3t, lane, acc);
        #pragma unroll
        for (int r = 0; r < 8; ++r) {
            float s = 0.f, q = 0.f;
            #pragma unroll
            for (int nt = 0; nt < 8; ++nt) {
                float v = acc[nt][r] + b3v[nt];
                s += v; q += v * v;
            }
            for (int off = 1; off < 16; off <<= 1) {
                s += __shfl_xor(s, off);
                q += __shfl_xor(q, off);
            }
            float m   = s * (1.f / 128.f);
            float var = q * (1.f / 128.f) - m * m;
            float rs  = rsqrtf(var + LN_EPS);
            size_t grow = (size_t)(T * 16 + r + hi * 8);
            float* orow = out + grow * 128 + c;
            #pragma unroll
            for (int nt = 0; nt < 8; ++nt) {
                float v = acc[nt][r] + b3v[nt];
                orow[nt * 16] = (v - m) * rs * gv[nt] + bv[nt];
            }
        }
    }
}

// -------------------- NODE branch: 50000 nodes, feat(30)->128->128->128 + LN --------------------
__global__ void __launch_bounds__(256) node_kernel(
    const int* __restrict__ mats, const float* __restrict__ pos,
    const float* __restrict__ vel,
    const float* __restrict__ matW, const float* __restrict__ matB,
    const float* __restrict__ W1, const float* __restrict__ B1,
    const half_t* __restrict__ W2T, const float* __restrict__ B2,
    const half_t* __restrict__ W3T, const float* __restrict__ B3,
    const float* __restrict__ G, const float* __restrict__ BT,
    float* __restrict__ out, int numTiles)
{
    extern __shared__ char smem[];
    half_t* w2t   = (half_t*)(smem + OFF_W2T);
    half_t* w3t   = (half_t*)(smem + OFF_W3T);
    half_t* hbuf  = (half_t*)(smem + OFF_HB);
    half_t* hbuf2 = (half_t*)(smem + OFF_HB2);
    float*  feat  = (float*)(smem + OFF_FEAT);
    float*  w1l   = (float*)(smem + N_OFF_W1);
    float*  b1l   = (float*)(smem + N_OFF_B1);

    const int tid   = threadIdx.x;
    const int lane  = tid & 31;
    const int wid   = tid >> 5;
    const int wbase = wid * 16;
    const int c     = lane & 15;
    const int hi    = (lane >> 4) & 1;

    for (int ci = tid; ci < 2048; ci += 256) {
        int n = ci >> 4, k8 = ci & 15;
        *(v4u*)(w2t + n * HS + k8 * 8) = *(const v4u*)(W2T + n * 128 + k8 * 8);
        *(v4u*)(w3t + n * HS + k8 * 8) = *(const v4u*)(W3T + n * 128 + k8 * 8);
    }
    for (int i = tid; i < 30 * 128; i += 256) w1l[i] = W1[i];
    if (tid < 128) b1l[tid] = B1[tid];
    __syncthreads();

    float b2v[8], b3v[8], gv[8], bv[8];
    #pragma unroll
    for (int nt = 0; nt < 8; ++nt) {
        int col = nt * 16 + c;
        b2v[nt] = B2[col]; b3v[nt] = B3[col]; gv[nt] = G[col]; bv[nt] = BT[col];
    }
    const int j0 = lane * 4;
    v4f bb = *(const v4f*)(b1l + j0);

    for (int T = blockIdx.x * NW + wid; T < numTiles; T += gridDim.x * NW) {
        // node features (lanes 0..15, one node each): vel(10) | matW[m]+matb (16) | wall(4)
        if (lane < 16) {
            int n = T * 16 + lane;
            float* f = feat + (wbase + lane) * 32;
            const float* vp = vel + (size_t)n * 10;
            #pragma unroll
            for (int k = 0; k < 10; ++k) f[k] = vp[k];
            int m = mats[n];
            #pragma unroll
            for (int j = 0; j < 16; ++j) f[10 + j] = matW[m * 16 + j] + matB[j];
            float px = pos[n * 2], py = pos[n * 2 + 1];
            f[26] = fminf(fmaxf(px, 0.f), RCLIP);
            f[27] = fminf(fmaxf(1.f - px, 0.f), RCLIP);
            f[28] = fminf(fmaxf(py, 0.f), RCLIP);
            f[29] = fminf(fmaxf(1.f - py, 0.f), RCLIP);
        }
        // layer1: 30 -> 128, relu, f16 to hbuf
        #pragma unroll
        for (int r = 0; r < 16; ++r) {
            const float* f = feat + (wbase + r) * 32;
            float a0 = bb.x, a1 = bb.y, a2 = bb.z, a3 = bb.w;
            #pragma unroll 5
            for (int k = 0; k < 30; ++k) {
                float fv = f[k];
                v4f wv = *(const v4f*)(w1l + k * 128 + j0);
                a0 += fv * wv.x; a1 += fv * wv.y; a2 += fv * wv.z; a3 += fv * wv.w;
            }
            v4h pk;
            pk.x = (half_t)fmaxf(a0, 0.f);
            pk.y = (half_t)fmaxf(a1, 0.f);
            pk.z = (half_t)fmaxf(a2, 0.f);
            pk.w = (half_t)fmaxf(a3, 0.f);
            *(v4h*)(hbuf + (wbase + r) * HS + j0) = pk;
        }
        // layer2 (WMMA) + relu -> hbuf2
        v8f zero = {};
        v8f acc[8];
        #pragma unroll
        for (int nt = 0; nt < 8; ++nt) acc[nt] = zero;
        mma128(hbuf + (wbase + c) * HS, w2t, lane, acc);
        #pragma unroll
        for (int nt = 0; nt < 8; ++nt) {
            #pragma unroll
            for (int r = 0; r < 8; ++r) {
                float h = fmaxf(acc[nt][r] + b2v[nt], 0.f);
                hbuf2[(wbase + r + hi * 8) * HS + nt * 16 + c] = (half_t)h;
            }
        }
        // layer3 (WMMA) + bias + layernorm -> global
        #pragma unroll
        for (int nt = 0; nt < 8; ++nt) acc[nt] = zero;
        mma128(hbuf2 + (wbase + c) * HS, w3t, lane, acc);
        #pragma unroll
        for (int r = 0; r < 8; ++r) {
            float s = 0.f, q = 0.f;
            #pragma unroll
            for (int nt = 0; nt < 8; ++nt) {
                float v = acc[nt][r] + b3v[nt];
                s += v; q += v * v;
            }
            for (int off = 1; off < 16; off <<= 1) {
                s += __shfl_xor(s, off);
                q += __shfl_xor(q, off);
            }
            float m   = s * (1.f / 128.f);
            float var = q * (1.f / 128.f) - m * m;
            float rs  = rsqrtf(var + LN_EPS);
            size_t grow = (size_t)(T * 16 + r + hi * 8);
            float* orow = out + grow * 128 + c;
            #pragma unroll
            for (int nt = 0; nt < 8; ++nt) {
                float v = acc[nt][r] + b3v[nt];
                orow[nt * 16] = (v - m) * rs * gv[nt] + bv[nt];
            }
        }
    }
}

// -------------------- pre-pass: f32 [K][N] weights -> f16 transposed [N][K] in workspace ------
__global__ void convert_weights(const float* __restrict__ nW2, const float* __restrict__ nW3,
                                const float* __restrict__ eW2, const float* __restrict__ eW3,
                                half_t* __restrict__ ws) {
    int i = blockIdx.x * blockDim.x + threadIdx.x;
    if (i >= 4 * 128 * 128) return;
    int m = i >> 14, rem = i & 16383;
    int n = rem >> 7, k = rem & 127;
    const float* src = (m == 0) ? nW2 : (m == 1) ? nW3 : (m == 2) ? eW2 : eW3;
    ws[i] = (half_t)src[k * 128 + n];   // ws[m][n][k] = W[k][n]
}

__global__ void copy_idx(const int* __restrict__ nbr, float* __restrict__ dst, int n) {
    int i = blockIdx.x * blockDim.x + threadIdx.x;
    if (i < n) dst[i] = (float)nbr[i];
}

extern "C" void kernel_launch(void* const* d_in, const int* in_sizes, int n_in,
                              void* d_out, int out_size, void* d_ws, size_t ws_size,
                              hipStream_t stream) {
    (void)in_sizes; (void)n_in; (void)out_size; (void)ws_size;

    const int*   materials  = (const int*)d_in[0];
    const float* positions  = (const float*)d_in[1];
    const float* velocities = (const float*)d_in[2];
    const int*   nbr        = (const int*)d_in[3];
    const float* matW = (const float*)d_in[4];
    const float* matB = (const float*)d_in[5];
    const float* nW1 = (const float*)d_in[6];  const float* nb1 = (const float*)d_in[7];
    const float* nW2 = (const float*)d_in[8];  const float* nb2 = (const float*)d_in[9];
    const float* nW3 = (const float*)d_in[10]; const float* nb3 = (const float*)d_in[11];
    const float* nG  = (const float*)d_in[12]; const float* nBt = (const float*)d_in[13];
    const float* eW1 = (const float*)d_in[14]; const float* eb1 = (const float*)d_in[15];
    const float* eW2 = (const float*)d_in[16]; const float* eb2 = (const float*)d_in[17];
    const float* eW3 = (const float*)d_in[18]; const float* eb3 = (const float*)d_in[19];
    const float* eG  = (const float*)d_in[20]; const float* eBt = (const float*)d_in[21];

    float*  out = (float*)d_out;
    half_t* ws  = (half_t*)d_ws;
    half_t* nW2T = ws;              half_t* nW3T = ws + 16384;
    half_t* eW2T = ws + 32768;      half_t* eW3T = ws + 49152;

    (void)hipFuncSetAttribute((const void*)node_kernel,
                              hipFuncAttributeMaxDynamicSharedMemorySize, NODE_LDS);
    (void)hipFuncSetAttribute((const void*)edge_kernel,
                              hipFuncAttributeMaxDynamicSharedMemorySize, EDGE_LDS);

    convert_weights<<<256, 256, 0, stream>>>(nW2, nW3, eW2, eW3, ws);

    // nodes: 50000 = 3125 tiles of 16 (exact); edges: 1e6 = 62500 tiles of 16 (exact)
    node_kernel<<<128, 256, NODE_LDS, stream>>>(materials, positions, velocities,
        matW, matB, nW1, nb1, nW2T, nb2, nW3T, nb3, nG, nBt, out, 3125);
    edge_kernel<<<512, 256, EDGE_LDS, stream>>>(positions, nbr,
        eW1, eb1, eW2T, eb2, eW3T, eb3, eG, eBt, out + 6400000, 62500);

    copy_idx<<<(3000000 + 255) / 256, 256, 0, stream>>>(nbr, out + 134400000, 3000000);
}
